// InfiniAttention_43095701848253
// MI455X (gfx1250) — compile-verified, matrix operand order checked by reading, not run on hardware
//
#include <hip/hip_runtime.h>
#include <hip/hip_bf16.h>
#include <math.h>

// ---------------- problem constants ----------------
#define BB   2
#define SS   2048
#define HID  1024
#define NH   16
#define HD   64
#define SCALE 0.125f                 // HD^-0.5
#define CEXP  0.1803368801111244f    // SCALE * log2(e)

typedef __bf16 bf16_t;
typedef __attribute__((ext_vector_type(16))) __bf16 v16bf;
typedef __attribute__((ext_vector_type(8)))  __bf16 v8bf;
typedef __attribute__((ext_vector_type(8)))  float  v8f;
typedef __attribute__((ext_vector_type(4)))  float  v4f;
typedef __attribute__((ext_vector_type(4)))  unsigned v4u;
typedef __attribute__((ext_vector_type(8)))  unsigned v8u;

// element counts in workspace
constexpr size_t N_X    = (size_t)BB * SS * HID;     // 4,194,304
constexpr size_t N_W    = (size_t)HID * HID;         // 1,048,576
constexpr size_t N_QKV  = (size_t)BB * NH * SS * HD; // 4,194,304
constexpr size_t N_MEMT = (size_t)NH * HD * HD;      // 65,536

// ---------------- helpers ----------------
__device__ __forceinline__ float fast_exp2(float x) {
    return __builtin_amdgcn_exp2f(x);               // v_exp_f32 (hardware exp2)
}
__device__ __forceinline__ bf16_t f2bf(float f) {
    unsigned u = __builtin_bit_cast(unsigned, f);
    unsigned r = u + 0x7FFFu + ((u >> 16) & 1u);          // RNE
    return __builtin_bit_cast(bf16_t, (unsigned short)(r >> 16));
}
__device__ __forceinline__ float bf2f(bf16_t b) {
    unsigned short s = __builtin_bit_cast(unsigned short, b);
    return __builtin_bit_cast(float, ((unsigned)s) << 16);
}
// two floats -> packed bf16 pair: 2 adds + 1 v_perm_b32
__device__ __forceinline__ unsigned pack_bf16(float flo, float fhi) {
    unsigned lo = __builtin_bit_cast(unsigned, flo) + 0x8000u;  // round-half-up
    unsigned hi = __builtin_bit_cast(unsigned, fhi) + 0x8000u;
    return __builtin_amdgcn_perm(hi, lo, 0x07060302u);          // {hi[31:16],lo[31:16]}
}

// Load one 16x32 bf16 A-fragment (or mirror-layout B-fragment) for
// v_wmma_f32_16x16x32_bf16.  Per ISA 7.12.2: lane L (L<16) holds row M=L,
// K = {k0+0..7, k0+16..23}; lane L>=16 holds K = {k0+8..15, k0+24..31}.
__device__ __forceinline__ v16bf load_frag(const bf16_t* base, int row_stride,
                                           int row, int k0, int lane) {
    const bf16_t* p = base + (size_t)row * row_stride + k0 + ((lane >> 4) << 3);
    v8bf lo = *(const v8bf*)(p);
    v8bf hi = *(const v8bf*)(p + 16);
    return __builtin_shufflevector(lo, hi, 0,1,2,3,4,5,6,7,8,9,10,11,12,13,14,15);
}

__device__ __forceinline__ v8f wmma_bf16(v16bf a, v16bf b, v8f c) {
    return __builtin_amdgcn_wmma_f32_16x16x32_bf16(false, a, false, b,
                                                   (short)0, c, false, false);
}

// ---------------- kernel 1: fp32 -> bf16 convert (8 elems/thread) ----------
__global__ void cvt_kernel(const float* __restrict__ src, bf16_t* __restrict__ dst, int n) {
    int i = (blockIdx.x * blockDim.x + threadIdx.x) * 8;
    int stride = gridDim.x * blockDim.x * 8;
    for (; i < n; i += stride) {
        v4f a = *(const v4f*)(src + i);
        v4f b = *(const v4f*)(src + i + 4);
        v4u p;
        p[0] = pack_bf16(a[0], a[1]);
        p[1] = pack_bf16(a[2], a[3]);
        p[2] = pack_bf16(b[0], b[1]);
        p[3] = pack_bf16(b[2], b[3]);
        *(v4u*)(dst + i) = p;
    }
}

// memT[h][e][d] = memory[h][d][e]  (bf16), so B-fragments load contiguously
__global__ void memT_kernel(const float* __restrict__ mem, bf16_t* __restrict__ memT) {
    int i = blockIdx.x * blockDim.x + threadIdx.x;   // 16*64*64 = 65536 threads
    int h = i >> 12, rem = i & 4095, e = rem >> 6, d = rem & 63;
    memT[((size_t)h * HD + e) * HD + d] = f2bf(mem[((size_t)h * HD + d) * HD + e]);
}

// ---------------- kernel 2: QKV projection GEMM ----------------
// C[m,n] = sum_k x[m,k] * W[n,k].  One wave per 64x64 tile; z selects q/k/v.
// q,k stored [B,H,S,D]; v stored transposed [B,H,D,S] (packed 16B stores).
__global__ __launch_bounds__(128) void qkv_gemm_kernel(
        const bf16_t* __restrict__ xb,
        const bf16_t* __restrict__ wqb, const bf16_t* __restrict__ wkb,
        const bf16_t* __restrict__ wvb,
        bf16_t* __restrict__ qb, bf16_t* __restrict__ kb, bf16_t* __restrict__ vTb) {
    const int lane  = threadIdx.x & 31;
    const int wave  = threadIdx.x >> 5;
    const int m_base = blockIdx.x * 64;
    const int n_base = blockIdx.y * 256 + wave * 64;
    const int which  = blockIdx.z;
    const bf16_t* W = (which == 0) ? wqb : (which == 1) ? wkb : wvb;

    const int rsel  = lane & 15;
    const int rhalf = (lane >> 4) * 8;

    v8f acc[4][4] = {};
    for (int k0 = 0; k0 < HID; k0 += 32) {
        v16bf af[4], bf[4];
        #pragma unroll
        for (int mi = 0; mi < 4; ++mi) af[mi] = load_frag(xb, HID, m_base + mi*16 + rsel, k0, lane);
        #pragma unroll
        for (int ni = 0; ni < 4; ++ni) bf[ni] = load_frag(W,  HID, n_base + ni*16 + rsel, k0, lane);
        #pragma unroll
        for (int mi = 0; mi < 4; ++mi)
            #pragma unroll
            for (int ni = 0; ni < 4; ++ni)
                acc[mi][ni] = wmma_bf16(af[mi], bf[ni], acc[mi][ni]);
    }
    if (which == 2) {
        // packed transposed store: consecutive i -> consecutive s
        #pragma unroll
        for (int mi = 0; mi < 4; ++mi)
            #pragma unroll
            for (int ni = 0; ni < 4; ++ni) {
                v4u pv;
                #pragma unroll
                for (int j = 0; j < 4; ++j)
                    pv[j] = pack_bf16(acc[mi][ni][2*j], acc[mi][ni][2*j + 1]);
                int row0 = m_base + mi*16 + rhalf;        // s start (aligned 8)
                int col  = n_base + ni*16 + rsel;
                int b = row0 >> 11, s0 = row0 & 2047;
                int h = col >> 6,   d  = col & 63;
                *(v4u*)&vTb[((size_t)(b * NH + h) * HD + d) * SS + s0] = pv;
            }
    } else {
        #pragma unroll
        for (int mi = 0; mi < 4; ++mi)
            #pragma unroll
            for (int ni = 0; ni < 4; ++ni)
                #pragma unroll
                for (int i = 0; i < 8; ++i) {
                    int row = m_base + mi*16 + rhalf + i;
                    int col = n_base + ni*16 + rsel;
                    int b = row >> 11, s = row & 2047;
                    int h = col >> 6,  d = col & 63;
                    bf16_t v = f2bf(acc[mi][ni][i]);
                    size_t bh = (size_t)(b * NH + h);
                    if (which == 0) qb[(bh * SS + s) * HD + d] = v;
                    else            kb[(bh * SS + s) * HD + d] = v;
                }
    }
}

// ---------------- kernel 3: causal flash attention (S^T trick) ----------------
// Scores computed transposed: S^T = K Q^T, so softmax stats for query s live
// in lane (s&15) (plus lane^16) -> 1 shuffle per reduction.  Scale folded
// into exp2: p = exp2(fma(s, CEXP, -m*CEXP)).  P packed into LDS [s][t]-major
// (exactly the P*V A-fragment layout).  Output a_localT fp32 [B,H,D,S].
__global__ __launch_bounds__(32) void attn_kernel(
        const bf16_t* __restrict__ qb, const bf16_t* __restrict__ kb,
        const bf16_t* __restrict__ vTb, float* __restrict__ a_localT) {
    __shared__ __align__(16) bf16_t ldsP[64 * 64];
    __shared__ __align__(16) float  sbc[64];
    const int lane  = threadIdx.x & 31;
    const int qt    = blockIdx.x;         // 0..31
    const int bh    = blockIdx.y;         // 0..31
    const int rsel  = lane & 15;
    const int rhalf = (lane >> 4) * 8;
    const int s_base = qt * 64;

    const bf16_t* qh = qb  + (size_t)bh * SS * HD;
    const bf16_t* kh = kb  + (size_t)bh * SS * HD;
    const bf16_t* vh = vTb + (size_t)bh * HD * SS;

    // Q as B-fragments (mirror loads from [s][d]-major), resident
    v16bf qf[4][2];
    #pragma unroll
    for (int si = 0; si < 4; ++si)
        #pragma unroll
        for (int kk = 0; kk < 2; ++kk)
            qf[si][kk] = load_frag(qh, HD, s_base + si*16 + rsel, kk*32, lane);

    v8f o[4][4] = {};                 // [si][di], rows = s, cols = d
    float mcol[4], lcol[4];           // raw-score stats for s = si*16 + rsel
    #pragma unroll
    for (int si = 0; si < 4; ++si) { mcol[si] = -INFINITY; lcol[si] = 0.f; }

    for (int tt = 0; tt <= qt; ++tt) {
        const int t_base = tt * 64;

        // prefetch next K/V tiles into cache (global_prefetch_b8)
        if (tt < qt) {
            __builtin_prefetch(kh + (size_t)(t_base + 64 + lane) * HD, 0, 3);
            __builtin_prefetch(kh + (size_t)(t_base + 96 + lane) * HD, 0, 3);
            __builtin_prefetch(vh + (size_t)lane        * SS + t_base + 64, 0, 3);
            __builtin_prefetch(vh + (size_t)(lane + 32) * SS + t_base + 64, 0, 3);
        }

        // ---- S^T = K Q^T : rows t, cols s (raw scores) ----
        v16bf kf[4][2];
        #pragma unroll
        for (int ti = 0; ti < 4; ++ti)
            #pragma unroll
            for (int kk = 0; kk < 2; ++kk)
                kf[ti][kk] = load_frag(kh, HD, t_base + ti*16 + rsel, kk*32, lane);
        v8f sc[4][4] = {};            // [ti][si]
        #pragma unroll
        for (int kk = 0; kk < 2; ++kk)
            #pragma unroll
            for (int ti = 0; ti < 4; ++ti)
                #pragma unroll
                for (int si = 0; si < 4; ++si)
                    sc[ti][si] = wmma_bf16(kf[ti][kk], qf[si][kk], sc[ti][si]);

        if (tt == qt) {               // causal mask on diagonal tile only
            #pragma unroll
            for (int ti = 0; ti < 4; ++ti)
                #pragma unroll
                for (int si = 0; si < 4; ++si)
                    #pragma unroll
                    for (int i = 0; i < 8; ++i)
                        if (ti*16 + rhalf + i > si*16 + rsel)
                            sc[ti][si][i] = -INFINITY;
        }

        // ---- online softmax per query column (scale folded into exp2) ----
        float alpha[4];
        #pragma unroll
        for (int si = 0; si < 4; ++si) {
            float tm = -INFINITY;
            #pragma unroll
            for (int ti = 0; ti < 4; ++ti)
                #pragma unroll
                for (int i = 0; i < 8; ++i) tm = fmaxf(tm, sc[ti][si][i]);
            tm = fmaxf(tm, __shfl_xor(tm, 16, 32));
            float nm = fmaxf(mcol[si], tm);
            alpha[si] = fast_exp2((mcol[si] - nm) * CEXP);
            mcol[si] = nm;
            float nmc = nm * CEXP;
            float rs = 0.f;
            #pragma unroll
            for (int ti = 0; ti < 4; ++ti)
                #pragma unroll
                for (int i = 0; i < 8; ++i) {
                    float p = fast_exp2(__builtin_fmaf(sc[ti][si][i], CEXP, -nmc));
                    sc[ti][si][i] = p;
                    rs += p;
                }
            rs += __shfl_xor(rs, 16, 32);
            lcol[si] = lcol[si] * alpha[si] + rs;
        }

        __syncthreads();
        if (lane < 16) {
            #pragma unroll
            for (int si = 0; si < 4; ++si) sbc[si*16 + rsel] = alpha[si];
        }
        // packed P store: 8 consecutive t for fixed s -> ds_store_b128
        #pragma unroll
        for (int ti = 0; ti < 4; ++ti)
            #pragma unroll
            for (int si = 0; si < 4; ++si) {
                v4u pv;
                #pragma unroll
                for (int j = 0; j < 4; ++j)
                    pv[j] = pack_bf16(sc[ti][si][2*j], sc[ti][si][2*j + 1]);
                *(v4u*)&ldsP[(si*16 + rsel) * 64 + ti*16 + rhalf] = pv;
            }
        __syncthreads();

        // rescale O by alpha (broadcast through LDS, packed float4 reads)
        #pragma unroll
        for (int si = 0; si < 4; ++si) {
            v4f a0 = *(const v4f*)&sbc[si*16 + rhalf];
            v4f a1 = *(const v4f*)&sbc[si*16 + rhalf + 4];
            #pragma unroll
            for (int di = 0; di < 4; ++di)
                #pragma unroll
                for (int i = 0; i < 8; ++i)
                    o[si][di][i] *= (i < 4) ? a0[i] : a1[i - 4];
        }

        // ---- O += P V ----
        #pragma unroll
        for (int kk = 0; kk < 2; ++kk) {
            v16bf pf[4], vf[4];
            #pragma unroll
            for (int si = 0; si < 4; ++si)
                pf[si] = load_frag(ldsP, 64, si*16 + rsel, kk*32, lane);
            #pragma unroll
            for (int di = 0; di < 4; ++di)
                vf[di] = load_frag(vh, SS, di*16 + rsel, t_base + kk*32, lane);
            #pragma unroll
            for (int si = 0; si < 4; ++si)
                #pragma unroll
                for (int di = 0; di < 4; ++di)
                    o[si][di] = wmma_bf16(pf[si], vf[di], o[si][di]);
        }
    }

    // finalize: broadcast 1/l, write a_localT[d][s] with packed 16B stores
    __syncthreads();
    if (lane < 16) {
        #pragma unroll
        for (int si = 0; si < 4; ++si) sbc[si*16 + rsel] = 1.f / lcol[si];
    }
    __syncthreads();
    float* outp = a_localT + (size_t)bh * HD * SS;
    #pragma unroll
    for (int si = 0; si < 4; ++si) {
        v4f r0 = *(const v4f*)&sbc[si*16 + rhalf];
        v4f r1 = *(const v4f*)&sbc[si*16 + rhalf + 4];
        int s0 = s_base + si*16 + rhalf;
        #pragma unroll
        for (int di = 0; di < 4; ++di) {
            int d = di*16 + rsel;
            v4f w0, w1;
            #pragma unroll
            for (int i = 0; i < 4; ++i) { w0[i] = o[si][di][i] * r0[i];
                                          w1[i] = o[si][di][i+4] * r1[i]; }
            *(v4f*)&outp[(size_t)d * SS + s0]     = w0;
            *(v4f*)&outp[(size_t)d * SS + s0 + 4] = w1;
        }
    }
}

// ---------------- kernel 4: memory retrieval + gate combine ----------------
__global__ __launch_bounds__(32) void combine_kernel(
        const bf16_t* __restrict__ qb, const float* __restrict__ a_localT,
        const bf16_t* __restrict__ memTb, const float* __restrict__ memnorm,
        const float* __restrict__ beta, bf16_t* __restrict__ comb) {
    __shared__ float snorm[64];
    const int lane  = threadIdx.x & 31;
    const int qt    = blockIdx.x;
    const int bh    = blockIdx.y;
    const int b     = bh >> 4, h = bh & 15;
    const int rsel  = lane & 15;
    const int rhalf = (lane >> 4) * 8;
    const int s_base = qt * 64;

    const float gate = 1.f / (1.f + __expf(-beta[h]));
    const bf16_t* qh = qb + (size_t)bh * SS * HD;
    const bf16_t* mh = memTb + (size_t)h * HD * HD;

    // sigma_q fragments (elu+1 elementwise, layout preserved) + norm dot
    v16bf sqf[4][2];
    float npart[4] = {0.f, 0.f, 0.f, 0.f};
    #pragma unroll
    for (int mi = 0; mi < 4; ++mi)
        #pragma unroll
        for (int kk = 0; kk < 2; ++kk) {
            v16bf q = load_frag(qh, HD, s_base + mi*16 + rsel, kk*32, lane);
            float sv[16];
            #pragma unroll
            for (int j = 0; j < 16; ++j) {
                float x = bf2f(q[j]);
                sv[j] = (x > 0.f) ? (x + 1.f) : __expf(x);   // elu(x)+1
                int d = kk*32 + ((j < 8) ? (rhalf + j) : (16 + rhalf + (j - 8)));
                npart[mi] += sv[j] * memnorm[h * HD + d];
            }
            v8u su;
            #pragma unroll
            for (int j2 = 0; j2 < 8; ++j2) su[j2] = pack_bf16(sv[2*j2], sv[2*j2 + 1]);
            sqf[mi][kk] = __builtin_bit_cast(v16bf, su);
        }
    #pragma unroll
    for (int mi = 0; mi < 4; ++mi) {
        float nrm = npart[mi] + __shfl_xor(npart[mi], 16, 32);
        nrm = fmaxf(nrm, 1e-6f);
        if (lane < 16) snorm[mi*16 + rsel] = nrm;
    }
    __syncthreads();

    // a_mem = sigma_q @ memory[h]
    v8f am[4][4] = {};
    #pragma unroll
    for (int kk = 0; kk < 2; ++kk) {
        v16bf mf[4];
        #pragma unroll
        for (int ni = 0; ni < 4; ++ni)
            mf[ni] = load_frag(mh, HD, ni*16 + rsel, kk*32, lane);
        #pragma unroll
        for (int mi = 0; mi < 4; ++mi)
            #pragma unroll
            for (int ni = 0; ni < 4; ++ni)
                am[mi][ni] = wmma_bf16(sqf[mi][kk], mf[ni], am[mi][ni]);
    }

    const float* alT = a_localT + (size_t)bh * SS * HD;   // [d][s]
    #pragma unroll
    for (int mi = 0; mi < 4; ++mi)
        #pragma unroll
        for (int ni = 0; ni < 4; ++ni) {
            int d   = ni*16 + rsel;
            int sr0 = s_base + mi*16 + rhalf;
            v4f l0 = *(const v4f*)&alT[(size_t)d * SS + sr0];
            v4f l1 = *(const v4f*)&alT[(size_t)d * SS + sr0 + 4];
            #pragma unroll
            for (int i = 0; i < 8; ++i) {
                int r  = mi*16 + rhalf + i;
                int sr = s_base + r;
                float amv = am[mi][ni][i] / snorm[r];
                float lv  = (i < 4) ? l0[i] : l1[i - 4];
                float c   = gate * amv + (1.f - gate) * lv;
                comb[(size_t)(b * SS + sr) * HID + h * HD + d] = f2bf(c);
            }
        }
}

// ---------------- kernel 5: output projection GEMM (fp32 out) ----------------
__global__ __launch_bounds__(128) void out_gemm_kernel(
        const bf16_t* __restrict__ cb, const bf16_t* __restrict__ wob,
        float* __restrict__ out) {
    const int lane  = threadIdx.x & 31;
    const int wave  = threadIdx.x >> 5;
    const int m_base = blockIdx.x * 64;
    const int n_base = blockIdx.y * 256 + wave * 64;
    const int rsel  = lane & 15;
    const int rhalf = (lane >> 4) * 8;

    v8f acc[4][4] = {};
    for (int k0 = 0; k0 < HID; k0 += 32) {
        v16bf af[4], bf[4];
        #pragma unroll
        for (int mi = 0; mi < 4; ++mi) af[mi] = load_frag(cb,  HID, m_base + mi*16 + rsel, k0, lane);
        #pragma unroll
        for (int ni = 0; ni < 4; ++ni) bf[ni] = load_frag(wob, HID, n_base + ni*16 + rsel, k0, lane);
        #pragma unroll
        for (int mi = 0; mi < 4; ++mi)
            #pragma unroll
            for (int ni = 0; ni < 4; ++ni)
                acc[mi][ni] = wmma_bf16(af[mi], bf[ni], acc[mi][ni]);
    }
    #pragma unroll
    for (int mi = 0; mi < 4; ++mi)
        #pragma unroll
        for (int ni = 0; ni < 4; ++ni)
            #pragma unroll
            for (int i = 0; i < 8; ++i) {
                int row = m_base + mi*16 + rhalf + i;
                int col = n_base + ni*16 + rsel;
                out[(size_t)row * HID + col] = acc[mi][ni][i];
            }
}

// ---------------- host launcher ----------------
extern "C" void kernel_launch(void* const* d_in, const int* in_sizes, int n_in,
                              void* d_out, int out_size, void* d_ws, size_t ws_size,
                              hipStream_t stream) {
    const float* hs      = (const float*)d_in[0];
    const float* wq      = (const float*)d_in[1];
    const float* wk      = (const float*)d_in[2];
    const float* wv      = (const float*)d_in[3];
    const float* wo      = (const float*)d_in[4];
    const float* beta    = (const float*)d_in[5];
    const float* memory  = (const float*)d_in[6];
    const float* memnorm = (const float*)d_in[7];
    float* out = (float*)d_out;

    // workspace layout (bf16 region first, fp32 a_localT after)
    bf16_t* base = (bf16_t*)d_ws;
    size_t off = 0;
    bf16_t* xb    = base + off; off += N_X;
    bf16_t* wqb   = base + off; off += N_W;
    bf16_t* wkb   = base + off; off += N_W;
    bf16_t* wvb   = base + off; off += N_W;
    bf16_t* wob   = base + off; off += N_W;
    bf16_t* qb    = base + off; off += N_QKV;
    bf16_t* kb    = base + off; off += N_QKV;
    bf16_t* vTb   = base + off; off += N_QKV;
    bf16_t* memTb = base + off; off += N_MEMT;
    bf16_t* combb = base + off; off += N_X;
    float*  a_localT = (float*)(base + off);     // N_QKV floats, [B,H,D,S]

    cvt_kernel<<<2048, 256, 0, stream>>>(hs, xb, (int)N_X);   // 8 elems/thread
    cvt_kernel<<< 512, 256, 0, stream>>>(wq, wqb, (int)N_W);
    cvt_kernel<<< 512, 256, 0, stream>>>(wk, wkb, (int)N_W);
    cvt_kernel<<< 512, 256, 0, stream>>>(wv, wvb, (int)N_W);
    cvt_kernel<<< 512, 256, 0, stream>>>(wo, wob, (int)N_W);
    memT_kernel<<<256, 256, 0, stream>>>(memory, memTb);

    qkv_gemm_kernel<<<dim3(64, 4, 3), 128, 0, stream>>>(xb, wqb, wkb, wvb, qb, kb, vTb);
    attn_kernel<<<dim3(32, 32), 32, 0, stream>>>(qb, kb, vTb, a_localT);
    combine_kernel<<<dim3(32, 32), 32, 0, stream>>>(qb, a_localT, memTb, memnorm, beta, combb);
    out_gemm_kernel<<<dim3(64, 4), 128, 0, stream>>>(combb, wob, out);
}
